// TinyDecoder_25357486916106
// MI455X (gfx1250) — matrix operand order repaired and checked by reading, not compiled, not run on hardware
//
#include <hip/hip_runtime.h>
#include <hip/hip_bf16.h>

// ---------------------------------------------------------------------------
// TinyDecoder on MI455X (gfx1250):
//   x  = tanh(c @ fc_w^T + fc_b)              (one-time, VALU kernel)
//   gi = x @ w_ih^T + b_ih                    (one-time, VALU kernel)
//   recurrence (dominant, 206 GFLOP):
//     per step: gh = h @ w_hh^T + b_hh  -> gates -> h_new, fused head output
//   Strategy: 16 persistent workgroups, one 16-row batch tile each (rows are
//   independent in the recurrence), everything LDS-resident, WMMA bf16 for
//   the per-step GEMM and the fused classification head.
//   GEMM loop is k-outer / n-inner: one LDS A-fragment feeds 3 WMMAs.
// ---------------------------------------------------------------------------

#define EMB   512
#define GATES 1536          // 3*EMB
#define NCLS  6
#define BT    16            // batch-tile rows per workgroup (one WMMA M tile)
#define GP    1540          // LDS float pitch for gi/gh (pad vs 1536: banks)
#define HP    520           // LDS bf16 pitch for h (pad vs 512: banks)
#define GRU_THREADS 1024    // 32 waves; 1536/16 = 96 n-tiles -> 3 per wave

typedef __attribute__((ext_vector_type(16))) __bf16 v16bf;
typedef __attribute__((ext_vector_type(8)))  __bf16 bf16x8;
typedef __attribute__((ext_vector_type(8)))  float  v8f;

__device__ __forceinline__ v16bf make_frag(bf16x8 lo, bf16x8 hi) {
    v16bf r;
#pragma unroll
    for (int i = 0; i < 8; ++i) { r[i] = lo[i]; r[i + 8] = hi[i]; }
    return r;
}

// A operand (16x32 bf16), per ISA layout: lanes 0-15 M=lane K=0..7/16..23,
// lanes 16-31 M=lane-16 K=8..15/24..31. Two 16B LDS loads per lane.
__device__ __forceinline__ v16bf ldA(const __bf16* h, int lm, int lh, int kt) {
    const __bf16* p = h + lm * HP + kt * 32 + lh * 8;
    return make_frag(*(const bf16x8*)p, *(const bf16x8*)(p + 16));
}

// B operand (32x16 bf16): lane n = lane%16, same K order as A
// (lanes 0-15: K=0..7/16..23, lanes 16-31: K=8..15/24..31). wrow = row n of a row-major
// [N,K] bf16 matrix: two 16B global loads.
__device__ __forceinline__ v16bf ldB(const __bf16* wrow, int lh, int kt) {
    const __bf16* p = wrow + kt * 32 + lh * 8;
    return make_frag(*(const bf16x8*)p, *(const bf16x8*)(p + 16));
}

__device__ __forceinline__ float sigm(float x) {
    return 1.0f / (1.0f + __expf(-x));
}

// --------------------------- one-time kernels ------------------------------

__global__ void convert_kernel(const float* __restrict__ w_hh,
                               const float* __restrict__ head_w,
                               __bf16* __restrict__ whh_b,
                               __bf16* __restrict__ head_pad_b) {
    int i = blockIdx.x * blockDim.x + threadIdx.x;
    if (i < GATES * EMB) whh_b[i] = (__bf16)w_hh[i];
    if (i < 16 * EMB) {          // head padded to 16 rows (N tile of 16)
        int r = i >> 9, k = i & (EMB - 1);
        head_pad_b[i] = (r < NCLS) ? (__bf16)head_w[r * EMB + k] : (__bf16)0.0f;
    }
}

__global__ __launch_bounds__(EMB) void fc_kernel(const float* __restrict__ c,
                                                 const float* __restrict__ w,
                                                 const float* __restrict__ b,
                                                 float* __restrict__ x) {
    int bb = blockIdx.x, e = threadIdx.x;
    const float4* cr = (const float4*)(c + (size_t)bb * EMB);
    const float4* wr = (const float4*)(w + (size_t)e * EMB);
    float acc = b[e];
#pragma unroll 4
    for (int k = 0; k < EMB / 4; ++k) {
        float4 a = cr[k], ww = wr[k];
        acc += a.x * ww.x + a.y * ww.y + a.z * ww.z + a.w * ww.w;
    }
    x[(size_t)bb * EMB + e] = tanhf(acc);
}

__global__ __launch_bounds__(EMB) void gi_kernel(const float* __restrict__ x,
                                                 const float* __restrict__ w,
                                                 const float* __restrict__ b,
                                                 float* __restrict__ gi) {
    int bb = blockIdx.x;
    const float4* xr = (const float4*)(x + (size_t)bb * EMB);
    for (int j = threadIdx.x; j < GATES; j += EMB) {
        const float4* wr = (const float4*)(w + (size_t)j * EMB);
        float acc = b[j];
#pragma unroll 4
        for (int k = 0; k < EMB / 4; ++k) {
            float4 a = xr[k], ww = wr[k];
            acc += a.x * ww.x + a.y * ww.y + a.z * ww.z + a.w * ww.w;
        }
        gi[(size_t)bb * GATES + j] = acc;
    }
}

// ----------------------- persistent GRU + head kernel ----------------------

__global__ __launch_bounds__(GRU_THREADS, 1)
void gru_kernel(const float*  __restrict__ gi_g,     // [256,1536] f32
                const __bf16* __restrict__ whh,      // [1536,512] bf16 row-major
                const float*  __restrict__ bhh,      // [1536]
                const __bf16* __restrict__ headw,    // [16,512] bf16 (padded)
                const float*  __restrict__ headb,    // [6]
                const int*    __restrict__ maxT_p,
                float* __restrict__ out)             // [256,T,6]
{
    extern __shared__ char smem[];
    float*  gi_s  = (float*)smem;                    // 16 x GP
    float*  gh_s  = gi_s + BT * GP;                  // 16 x GP
    float*  hf    = gh_s + BT * GP;                  // 16 x 512  (f32 state)
    float*  bhh_s = hf + BT * EMB;                   // 1536
    __bf16* hb    = (__bf16*)(bhh_s + GATES);        // 16 x HP   (bf16 mirror)

    const int T    = *maxT_p;
    const int tid  = threadIdx.x;
    const int bBase = blockIdx.x * BT;
    const int lane = tid & 31, wave = tid >> 5;
    const int lh = lane >> 4, lm = lane & 15;

    // stage gi tile, b_hh, init h = 0
    for (int i = tid; i < BT * GATES; i += GRU_THREADS) {
        int r = i / GATES, col = i - r * GATES;
        gi_s[r * GP + col] = gi_g[(size_t)(bBase + r) * GATES + col];
    }
    for (int i = tid; i < GATES; i += GRU_THREADS) bhh_s[i] = bhh[i];
    for (int i = tid; i < BT * EMB; i += GRU_THREADS) hf[i] = 0.0f;
    for (int i = tid; i < BT * HP; i += GRU_THREADS) hb[i] = (__bf16)0.0f;

    const float hbias = (lm < NCLS) ? headb[lm] : 0.0f;   // head bias per lane

    // per-lane output columns for this wave's 3 n-tiles
    const int n0 = wave * 48 + lm;                   // wave*3 tiles * 16 cols
    const __bf16* w0 = whh + (size_t)n0 * EMB;
    const __bf16* w1 = w0 + (size_t)16 * EMB;
    const __bf16* w2 = w0 + (size_t)32 * EMB;

    __syncthreads();

    const float bias0 = bhh_s[n0];
    const float bias1 = bhh_s[n0 + 16];
    const float bias2 = bhh_s[n0 + 32];

    for (int t = 0; t < T; ++t) {
        // --- gh = h @ w_hh^T + b_hh : M=16, N=1536, K=512 -----------------
        // k-outer / n-inner: one LDS A-fragment feeds 3 WMMAs (B from L2).
        v8f acc0, acc1, acc2;
#pragma unroll
        for (int v = 0; v < 8; ++v) { acc0[v] = bias0; acc1[v] = bias1; acc2[v] = bias2; }
#pragma unroll 4
        for (int kt = 0; kt < EMB / 32; ++kt) {
            v16bf a = ldA(hb, lm, lh, kt);
            acc0 = __builtin_amdgcn_wmma_f32_16x16x32_bf16(
                     false, a, false, ldB(w0, lh, kt), (short)0, acc0, false, false);
            acc1 = __builtin_amdgcn_wmma_f32_16x16x32_bf16(
                     false, a, false, ldB(w1, lh, kt), (short)0, acc1, false, false);
            acc2 = __builtin_amdgcn_wmma_f32_16x16x32_bf16(
                     false, a, false, ldB(w2, lh, kt), (short)0, acc2, false, false);
        }
        {
            float* g0 = gh_s + n0;                   // row stride GP
#pragma unroll
            for (int v = 0; v < 8; ++v) {
                float* gr = g0 + (v + 8 * lh) * GP;
                gr[0]  = acc0[v];
                gr[16] = acc1[v];
                gr[32] = acc2[v];
            }
        }
        __syncthreads();

        // --- gates + state update (r,z,n; PyTorch GRU) ---
        for (int i = tid; i < BT * EMB; i += GRU_THREADS) {
            const int row = i >> 9, e = i & (EMB - 1);
            const float* gir = gi_s + row * GP;
            const float* ghr = gh_s + row * GP;
            float r  = sigm(gir[e] + ghr[e]);
            float z  = sigm(gir[EMB + e] + ghr[EMB + e]);
            float nn = tanhf(gir[2 * EMB + e] + r * ghr[2 * EMB + e]);
            float hp = hf[row * EMB + e];
            float hn = (1.0f - z) * nn + z * hp;
            hf[row * EMB + e] = hn;
            hb[row * HP + e]  = (__bf16)hn;
        }
        __syncthreads();

        // --- fused head: out[b,t,:] = h_new @ head_w^T + head_b (wave 0) ---
        // Safe without a third barrier: wave 0 only READS hb here; other waves
        // proceed into the next step's GEMM which also only reads hb and
        // writes gh_s (already consumed before the barrier above).
        if (wave == 0) {
            const __bf16* wrow = headw + (size_t)lm * EMB;
            v8f acc;
#pragma unroll
            for (int v = 0; v < 8; ++v) acc[v] = hbias;
#pragma unroll 4
            for (int kt = 0; kt < EMB / 32; ++kt) {
                v16bf a = ldA(hb, lm, lh, kt);
                acc = __builtin_amdgcn_wmma_f32_16x16x32_bf16(
                        false, a, false, ldB(wrow, lh, kt), (short)0, acc, false, false);
            }
            if (lm < NCLS) {
#pragma unroll
                for (int v = 0; v < 8; ++v) {
                    const int m = v + 8 * lh;
                    out[((size_t)(bBase + m) * T + t) * NCLS + lm] = acc[v];
                }
            }
        }
    }
}

// ------------------------------- launcher ----------------------------------

extern "C" void kernel_launch(void* const* d_in, const int* in_sizes, int n_in,
                              void* d_out, int out_size, void* d_ws, size_t ws_size,
                              hipStream_t stream) {
    const float* c      = (const float*)d_in[0];
    const float* fc_w   = (const float*)d_in[1];
    const float* fc_b   = (const float*)d_in[2];
    const float* w_ih   = (const float*)d_in[3];
    const float* b_ih   = (const float*)d_in[4];
    const float* w_hh   = (const float*)d_in[5];
    const float* b_hh   = (const float*)d_in[6];
    const float* head_w = (const float*)d_in[7];
    const float* head_b = (const float*)d_in[8];
    const int*   maxT   = (const int*)d_in[9];
    float* out = (float*)d_out;

    // workspace layout (~3.6 MB total)
    char* ws = (char*)d_ws;
    float*  gi    = (float*)ws;                          // 256*1536 f32
    float*  x     = (float*)(ws + 256 * GATES * 4);      // 256*512  f32
    __bf16* whh_b = (__bf16*)(ws + 256 * GATES * 4 + 256 * EMB * 4);
    __bf16* hw_b  = (__bf16*)((char*)whh_b + (size_t)GATES * EMB * 2);

    convert_kernel<<<(GATES * EMB + 255) / 256, 256, 0, stream>>>(
        w_hh, head_w, whh_b, hw_b);
    fc_kernel<<<256, EMB, 0, stream>>>(c, fc_w, fc_b, x);
    gi_kernel<<<256, EMB, 0, stream>>>(x, w_ih, b_ih, gi);

    // dynamic LDS: gi(16xGP) + gh(16xGP) + hf(16x512) + bhh(1536) f32
    //            + hb(16xHP) bf16  ~= 247 KB  (<= 320 KB/WGP on gfx1250)
    size_t smem = (size_t)(BT * GP + BT * GP + BT * EMB + GATES) * sizeof(float)
                + (size_t)(BT * HP) * 2;
    gru_kernel<<<256 / BT, GRU_THREADS, smem, stream>>>(
        gi, whh_b, b_hh, hw_b, head_b, maxT, out);
}
